// AttentionalGCN_36790689858064
// MI455X (gfx1250) — matrix operand order repaired and checked by reading, not compiled, not run on hardware
//
#include <hip/hip_runtime.h>

typedef _Float16 f16;
typedef __attribute__((ext_vector_type(16))) _Float16 v16h;
typedef __attribute__((ext_vector_type(8)))  _Float16 v8h;
typedef __attribute__((ext_vector_type(8)))  float    v8f;
typedef __attribute__((ext_vector_type(4)))  float    v4f;
typedef __attribute__((ext_vector_type(4)))  unsigned int u32x4;
typedef __attribute__((ext_vector_type(8)))  unsigned int u32x8;

#define N_NODES 8192
#define DIM     64
#define LDSB    (64 * 40)          // padded f16 tile: 64 rows x 40 (80 B rows)

union V16 { v16h v; v8h h[2]; };

static __device__ __forceinline__ v8h cvt8(v4f a, v4f b) {
  v8h r;
  r[0]=(f16)a[0]; r[1]=(f16)a[1]; r[2]=(f16)a[2]; r[3]=(f16)a[3];
  r[4]=(f16)b[0]; r[5]=(f16)b[1]; r[6]=(f16)b[2]; r[7]=(f16)b[3];
  return r;
}

static __device__ __forceinline__ v8f wmma_f16(v16h a, v16h b, v8f c) {
  return __builtin_amdgcn_wmma_f32_16x16x32_f16(false, a, false, b, (short)0, c,
                                                false, false);
}

// ---------------------------------------------------------------------------
// Prep: fold W_obj + W_skip, transpose all weights to [n][k] in f16.
// ---------------------------------------------------------------------------
__global__ void k_prep(const float* __restrict__ Wobj, const float* __restrict__ Wnobj,
                       const float* __restrict__ Wrel, const float* __restrict__ Wskip,
                       f16* __restrict__ Wc_t, f16* __restrict__ Wrel_t,
                       f16* __restrict__ Wnobj_t) {
  int idx = blockIdx.x * 256 + threadIdx.x;
  if (idx < DIM * DIM) {
    int k = idx >> 6, n = idx & 63;
    Wc_t[n * DIM + k]    = (f16)(Wobj[k * DIM + n] + Wskip[k * DIM + n]);
    Wrel_t[n * DIM + k]  = (f16)(Wrel[k * DIM + n]);
    Wnobj_t[n * DIM + k] = (f16)(Wnobj[k * DIM + n]);
  }
}

// ---------------------------------------------------------------------------
// Stage 1: base = x@(Wobj+Wskip) + r@Wrel + biases -> out
//          Yt[n][row] = f16(x@Wnobj + b_nobj)      -> workspace (transposed!)
// One wave per 16 output rows. Grid: 64 x 256.
// ---------------------------------------------------------------------------
__global__ void __launch_bounds__(256) k_stage1(
    const float* __restrict__ x, const float* __restrict__ r,
    const float* __restrict__ b_obj, const float* __restrict__ b_nobj,
    const float* __restrict__ b_rel, const float* __restrict__ b_skip,
    const f16* __restrict__ Wc_t, const f16* __restrict__ Wrel_t,
    const f16* __restrict__ Wnobj_t,
    float* __restrict__ out, f16* __restrict__ Yt) {
  const int L   = threadIdx.x & 31;
  const int w   = threadIdx.x >> 5;
  const int r0  = (blockIdx.x * 8 + w) * 16;
  const int m16 = L & 15;
  const int hh  = L >> 4;

  v8f cb[4] = {};
  v8f cy[4] = {};

#pragma unroll
  for (int k0 = 0; k0 < DIM; k0 += 32) {
    const int kb = k0 + hh * 8;                 // A frag: K chunks [kb,+8) and +16
    V16 ax, ar;
    const float* xp = x + (r0 + m16) * DIM + kb;
    ax.h[0] = cvt8(*(const v4f*)(xp),      *(const v4f*)(xp + 4));
    ax.h[1] = cvt8(*(const v4f*)(xp + 16), *(const v4f*)(xp + 20));
    const float* rp = r + (r0 + m16) * DIM + kb;
    ar.h[0] = cvt8(*(const v4f*)(rp),      *(const v4f*)(rp + 4));
    ar.h[1] = cvt8(*(const v4f*)(rp + 16), *(const v4f*)(rp + 20));

    const int kb2 = k0 + hh * 8;                // B frag: K chunks [kb2,+8) and +16
#pragma unroll
    for (int nt = 0; nt < 4; ++nt) {
      const int n = nt * 16 + m16;
      V16 b;
      const f16* p = Wc_t + n * DIM + kb2;
      b.h[0] = *(const v8h*)p; b.h[1] = *(const v8h*)(p + 16);
      cb[nt] = wmma_f16(ax.v, b.v, cb[nt]);

      p = Wrel_t + n * DIM + kb2;
      b.h[0] = *(const v8h*)p; b.h[1] = *(const v8h*)(p + 16);
      cb[nt] = wmma_f16(ar.v, b.v, cb[nt]);

      p = Wnobj_t + n * DIM + kb2;
      b.h[0] = *(const v8h*)p; b.h[1] = *(const v8h*)(p + 16);
      cy[nt] = wmma_f16(ax.v, b.v, cy[nt]);
    }
  }

#pragma unroll
  for (int nt = 0; nt < 4; ++nt) {
    const int n    = nt * 16 + m16;
    const float bb = b_obj[n] + b_skip[n] + b_rel[n];
    const float bn = b_nobj[n];
    v8h yv;
#pragma unroll
    for (int g = 0; g < 8; ++g) {
      const int row = r0 + hh * 8 + g;          // C/D: VGPR g -> M = g + 8*hh
      out[row * DIM + n] = cb[nt][g] + bb;
      yv[g] = (f16)(cy[nt][g] + bn);
    }
    // transposed store: Yt[n][r0+hh*8 .. +8) is contiguous -> one b128
    *(v8h*)(Yt + (size_t)n * N_NODES + r0 + hh * 8) = yv;
  }
}

// ---------------------------------------------------------------------------
// Stage 2: out += A.T @ Y  (HBM-bound 8192x8192x64 GEMM)
// Block: 256 thr (8 waves) owns 64 output rows; K tiled by 32; double-buffered
// LDS; B tile DMA'd by the Tensor Data Mover with HW row padding.
// Grid: 128 x 256.
// ---------------------------------------------------------------------------
__global__ void __launch_bounds__(256) k_stage2(
    const float* __restrict__ A, const f16* __restrict__ Yt,
    float* __restrict__ out) {
  __shared__ alignas(16) f16 lA[2][LDSB];   // [m][k], 40-f16 row stride
  __shared__ alignas(16) f16 lB[2][LDSB];   // [n][k], 40-f16 row stride (TDM pad)

  const int t    = threadIdx.x;
  const int L    = t & 31;
  const int w    = t >> 5;
  const int i0   = blockIdx.x * 64;
  const int m16  = L & 15;
  const int hh   = L >> 4;
  const int mt   = w & 3;
  const int nh   = w >> 2;
  const int mloc = mt * 16 + m16;
  const int n0   = nh * 32 + m16;

  v8f c0 = {}, c1 = {};

  // A tile: manual (needs f32->f16 convert + transpose), nontemporal stream.
  auto stageA = [&](int j0, int buf) {
#pragma unroll
    for (int p = 0; p < 8; ++p) {
      const int q = p * 256 + t;
      const int k = q >> 6, m = q & 63;
      const float av = __builtin_nontemporal_load(A + (j0 + k) * N_NODES + i0 + m);
      lA[buf][m * 40 + k] = (f16)av;
    }
  };

  // B tile: TDM 2D copy Yt[0..63][j0..j0+32) -> lB[buf] with HW padding:
  // rows of 64 B (tile_dim0=32 f16), pad 16 B each (interval=16 dw, amount=4 dw)
  // -> 80 B LDS row stride, matching the ds_load_b128 fragment reads.
  auto issueTDM = [&](int j0, int buf) {
    if (w == 0) {
      const unsigned long long ga = (unsigned long long)(size_t)(Yt + j0);
      u32x4 g0;
      g0[0] = 1u;                                            // count=1
      g0[1] = (unsigned)(size_t)(&lB[buf][0]);               // lds_addr (bytes)
      g0[2] = (unsigned)ga;                                  // global_addr[31:0]
      g0[3] = (unsigned)((ga >> 32) & 0x01FFFFFFu) | (2u << 30); // addr[56:32]|type=2
      u32x8 g1;
      g1[0] = (1u << 16)        // data_size = 2 bytes
            | (1u << 20)        // pad_enable
            | (3u << 22)        // pad_interval: 16 DWORDs (64 B)
            | (3u << 25);       // pad_amount: 4 DWORDs (16 B)
      g1[1] = (8192u & 0xFFFFu) << 16;                 // tensor_dim0[15:0]
      g1[2] = ((8192u >> 16) & 0xFFFFu) | (64u << 16); // dim0 hi | tensor_dim1=64
      g1[3] = (32u << 16);                             // tile_dim0 = 32 elements
      g1[4] = 64u;                                     // tile_dim1 = 64, tile_dim2=0
      g1[5] = 8192u;                                   // tensor_dim0_stride[31:0]
      g1[6] = 0u;
      g1[7] = 0u;
      asm volatile("tensor_load_to_lds %0, %1, null, null"
                   :: "s"(g0), "s"(g1) : "memory");
    }
  };

  stageA(0, 0);
  issueTDM(0, 0);

  for (int jt = 0; jt < N_NODES / 32; ++jt) {
    const int buf = jt & 1;
    if (w == 0) __builtin_amdgcn_s_wait_tensorcnt(0);  // TDM(jt) landed in LDS
    __syncthreads();                                   // tile jt published; buf^1 free
    if (jt + 1 < N_NODES / 32) {
      stageA((jt + 1) * 32, buf ^ 1);                  // overlap with WMMAs below
      issueTDM((jt + 1) * 32, buf ^ 1);
    }

    V16 a, b0, b1;
    const f16* ap = &lA[buf][mloc * 40 + hh * 8];
    a.h[0] = *(const v8h*)ap;
    a.h[1] = *(const v8h*)(ap + 16);
    const f16* bp0 = &lB[buf][n0 * 40 + hh * 8];
    b0.h[0] = *(const v8h*)bp0;
    b0.h[1] = *(const v8h*)(bp0 + 16);
    const f16* bp1 = &lB[buf][(n0 + 16) * 40 + hh * 8];
    b1.h[0] = *(const v8h*)bp1;
    b1.h[1] = *(const v8h*)(bp1 + 16);

    c0 = wmma_f16(a.v, b0.v, c0);
    c1 = wmma_f16(a.v, b1.v, c1);
  }

#pragma unroll
  for (int g = 0; g < 8; ++g) {
    const int row = i0 + mt * 16 + hh * 8 + g;
    out[row * DIM + n0]      += c0[g];
    out[row * DIM + n0 + 16] += c1[g];
  }
}

// ---------------------------------------------------------------------------
extern "C" void kernel_launch(void* const* d_in, const int* in_sizes, int n_in,
                              void* d_out, int out_size, void* d_ws, size_t ws_size,
                              hipStream_t stream) {
  (void)in_sizes; (void)n_in; (void)out_size; (void)ws_size;
  const float* x      = (const float*)d_in[0];
  const float* r      = (const float*)d_in[1];
  const float* A      = (const float*)d_in[2];
  const float* W_obj  = (const float*)d_in[3];
  const float* b_obj  = (const float*)d_in[4];
  const float* W_nobj = (const float*)d_in[5];
  const float* b_nobj = (const float*)d_in[6];
  const float* W_rel  = (const float*)d_in[7];
  const float* b_rel  = (const float*)d_in[8];
  const float* W_skip = (const float*)d_in[9];
  const float* b_skip = (const float*)d_in[10];
  // d_in[11]/d_in[12] (Wa_w, Wa_b): softmax rows sum to 1, so
  // einsum('ij,ik->ik', attn, agg) == agg exactly -> attention branch vanishes.
  float* out = (float*)d_out;

  f16* Yt      = (f16*)d_ws;                    // [DIM][N_NODES] f16 = 1 MB
  f16* Wc_t    = Yt + (size_t)N_NODES * DIM;
  f16* Wrel_t  = Wc_t + DIM * DIM;
  f16* Wnobj_t = Wrel_t + DIM * DIM;

  k_prep<<<16, 256, 0, stream>>>(W_obj, W_nobj, W_rel, W_skip, Wc_t, Wrel_t, Wnobj_t);
  k_stage1<<<64, 256, 0, stream>>>(x, r, b_obj, b_nobj, b_rel, b_skip,
                                   Wc_t, Wrel_t, Wnobj_t, out, Yt);
  k_stage2<<<128, 256, 0, stream>>>(A, Yt, out);
}
